// SparseMoE_23021024707547
// MI455X (gfx1250) — matrix operand order repaired and checked by reading, not compiled, not run on hardware
//
#include <hip/hip_runtime.h>
#include <hip/hip_bf16.h>
#include <math.h>

#define D  1024
#define E  8
#define TM 32

typedef __attribute__((ext_vector_type(16))) __bf16 bf16x16;
typedef __attribute__((ext_vector_type(8)))  __bf16 bf16x8;
typedef __attribute__((ext_vector_type(8)))  float  f32x8;

__device__ __forceinline__ unsigned short f2bf_bits(float f) {
    unsigned u = __builtin_bit_cast(unsigned, f);
    u += 0x7fffu + ((u >> 16) & 1u);          // round-to-nearest-even
    return (unsigned short)(u >> 16);
}

// A-matrix 16x32 bf16 fragment from row-major LDS tile [TM][D]:
// lanes 0-15: M=row, K = k0+0..7 (v0-3), k0+16..23 (v4-7); lanes 16-31: K=+8
__device__ __forceinline__ bf16x16 load_a_frag(const unsigned short* Xs, int row, int khalf, int k0) {
    const bf16x8* p0 = (const bf16x8*)(Xs + row * D + k0 + khalf * 8);
    const bf16x8* p1 = (const bf16x8*)(Xs + row * D + k0 + 16 + khalf * 8);
    bf16x8 lo = *p0, hi = *p1;
    return __builtin_shufflevector(lo, hi, 0,1,2,3,4,5,6,7,8,9,10,11,12,13,14,15);
}

// B-matrix 32x16 bf16 fragment from transposed bf16 weights Wt[n][k]:
// lane holds column n, K = kk+0..7 and kk+16..23 (kk includes lane-half*8)
__device__ __forceinline__ bf16x16 load_b_frag(const unsigned short* Wte, int n, int kk) {
    const bf16x8* p = (const bf16x8*)(Wte + (size_t)n * D + kk);
    bf16x8 lo = p[0], hi = p[2];
    return __builtin_shufflevector(lo, hi, 0,1,2,3,4,5,6,7,8,9,10,11,12,13,14,15);
}

// ---------------------------------------------------------------- init
__global__ void moe_init_kernel(int* counts, int* fill) {
    int t = threadIdx.x;
    if (t < E) { counts[t] = 0; fill[t] = 0; }
}

// ---------------------------------------------------------------- weight transpose-convert: Wt[e][n][k] = bf16(W[e][k][n])
__global__ __launch_bounds__(256) void moe_wcvt_kernel(const float* __restrict__ W,
                                                       unsigned short* __restrict__ Wt)
{
    __shared__ float tile[32][33];
    const int e  = blockIdx.z;
    const int kb = blockIdx.x * 32;
    const int nb = blockIdx.y * 32;
    const float* __restrict__ src = W + (size_t)e * D * D;
    unsigned short* __restrict__ dst = Wt + (size_t)e * D * D;
    const int tx = threadIdx.x;   // 0..31
    const int ty = threadIdx.y;   // 0..7
#pragma unroll
    for (int i = ty; i < 32; i += 8)
        tile[i][tx] = src[(size_t)(kb + i) * D + nb + tx];
    __syncthreads();
#pragma unroll
    for (int i = ty; i < 32; i += 8)
        dst[(size_t)(nb + i) * D + kb + tx] = f2bf_bits(tile[tx][i]);
}

// ---------------------------------------------------------------- gating: one wave32 per token
__global__ __launch_bounds__(256) void moe_gate_kernel(
    const float* __restrict__ x, const float* __restrict__ Wg, const float* __restrict__ bg,
    int* __restrict__ assign, float* __restrict__ gatew, int* __restrict__ counts, int NT)
{
    const int wid  = (int)((blockIdx.x * 256u + threadIdx.x) >> 5);
    const int lane = threadIdx.x & 31;
    if (wid >= NT) return;
    const float* __restrict__ xr = x + (size_t)wid * D;

    float acc[E];
#pragma unroll
    for (int j = 0; j < E; ++j) acc[j] = 0.0f;

    for (int d = lane; d < D; d += 32) {
        const float xv = xr[d];
        const float4* wr = (const float4*)(Wg + (size_t)d * E);
        float4 w0 = wr[0], w1 = wr[1];
        acc[0] += xv * w0.x; acc[1] += xv * w0.y; acc[2] += xv * w0.z; acc[3] += xv * w0.w;
        acc[4] += xv * w1.x; acc[5] += xv * w1.y; acc[6] += xv * w1.z; acc[7] += xv * w1.w;
    }
#pragma unroll
    for (int j = 0; j < E; ++j) {
        float v = acc[j];
#pragma unroll
        for (int off = 16; off > 0; off >>= 1) v += __shfl_xor(v, off, 32);
        acc[j] = v;
    }
    if (lane == 0) {
        float mx = -1e30f; int a = 0;
#pragma unroll
        for (int j = 0; j < E; ++j) { float l = acc[j] + bg[j]; acc[j] = l; if (l > mx) { mx = l; a = j; } }
        float s = 0.0f;
#pragma unroll
        for (int j = 0; j < E; ++j) s += expf(acc[j] - mx);
        assign[wid] = a;
        gatew[wid]  = 1.0f / s;               // softmax prob at its own argmax
        atomicAdd(&counts[a], 1);
    }
}

// ---------------------------------------------------------------- scan: padded segments + tile tables
__global__ void moe_scan_kernel(const int* __restrict__ counts, int* __restrict__ seg_base,
                                int* __restrict__ tile_expert, int* __restrict__ tile_base,
                                int* __restrict__ n_tiles, int* __restrict__ sorted)
{
    if (threadIdx.x != 0 || blockIdx.x != 0) return;
    int base = 0, tile = 0;
    for (int e = 0; e < E; ++e) {
        seg_base[e] = base;
        int c  = counts[e];
        int nt = (c + TM - 1) / TM;
        for (int i = 0; i < nt; ++i) { tile_expert[tile] = e; tile_base[tile] = base + i * TM; ++tile; }
        for (int i = c; i < nt * TM; ++i) sorted[base + i] = -1;   // padding rows
        base += nt * TM;
    }
    n_tiles[0] = tile;
}

// ---------------------------------------------------------------- scatter tokens into segments
__global__ __launch_bounds__(256) void moe_scatter_kernel(
    const int* __restrict__ assign, const int* __restrict__ seg_base,
    int* __restrict__ fill, int* __restrict__ sorted, int NT)
{
    int t = blockIdx.x * 256 + threadIdx.x;
    if (t >= NT) return;
    int a   = assign[t];
    int pos = atomicAdd(&fill[a], 1);
    sorted[seg_base[a] + pos] = t;
}

#define WMMA_BF16(A, B, C) __builtin_amdgcn_wmma_f32_16x16x32_bf16(false, (A), false, (B), (short)0, (C), false, false)

// ---------------------------------------------------------------- fused 2-GEMM expert MLP, bf16 WMMA
__global__ __launch_bounds__(256) void moe_mlp_kernel(
    const float* __restrict__ x,
    const unsigned short* __restrict__ Wt1, const float* __restrict__ b1,
    const unsigned short* __restrict__ Wt2, const float* __restrict__ b2,
    const int* __restrict__ sorted, const int* __restrict__ tile_expert,
    const int* __restrict__ tile_base, const int* __restrict__ n_tiles,
    const float* __restrict__ gate_w, float* __restrict__ out)
{
    __shared__ unsigned short Xs[TM * D];   // 64 KB: X tile (bf16), then reused as H tile

    const int tile = blockIdx.x;
    if (tile >= n_tiles[0]) return;
    const int e    = tile_expert[tile];
    const int base = tile_base[tile];
    const int tid  = threadIdx.x;

    // stage X tile (fp32 -> bf16) into LDS; padding rows (token -1) are zero
    for (int i = tid; i < TM * D; i += 256) {
        int r = i >> 10;
        int c = i & (D - 1);
        int t = sorted[base + r];
        Xs[i] = f2bf_bits(t >= 0 ? x[(size_t)t * D + c] : 0.0f);
    }
    __syncthreads();

    const int lane    = tid & 31;
    const int wave    = tid >> 5;
    const int nlo     = lane & 15;
    const int khalf   = lane >> 4;
    const int colbase = wave * 128;          // 8 waves x 128 cols = 1024

    const unsigned short* __restrict__ Wt1e = Wt1 + (size_t)e * D * D;
    const unsigned short* __restrict__ Wt2e = Wt2 + (size_t)e * D * D;

    const f32x8 vzero = {0.f,0.f,0.f,0.f,0.f,0.f,0.f,0.f};
    f32x8 acc[2][8];
#pragma unroll
    for (int m = 0; m < 2; ++m)
#pragma unroll
        for (int t8 = 0; t8 < 8; ++t8) acc[m][t8] = vzero;

    // ---- phase 1: H = gelu(X @ W1[e] + b1[e]) ----
    for (int k0 = 0; k0 < D; k0 += 32) {
        bf16x16 a0 = load_a_frag(Xs, nlo,      khalf, k0);
        bf16x16 a1 = load_a_frag(Xs, nlo + 16, khalf, k0);
        const int kk = k0 + khalf * 8;
#pragma unroll
        for (int t8 = 0; t8 < 8; ++t8) {
            const int n = colbase + t8 * 16 + nlo;
            bf16x16 b = load_b_frag(Wt1e, n, kk);
            acc[0][t8] = WMMA_BF16(a0, b, acc[0][t8]);
            acc[1][t8] = WMMA_BF16(a1, b, acc[1][t8]);
        }
    }
    __syncthreads();    // all waves done reading X tile

    // bias + exact GELU, write H (bf16) over the X tile
#pragma unroll
    for (int m = 0; m < 2; ++m)
#pragma unroll
    for (int t8 = 0; t8 < 8; ++t8) {
        const int   n    = colbase + t8 * 16 + nlo;
        const float bias = b1[e * D + n];
#pragma unroll
        for (int v = 0; v < 8; ++v) {
            const int row = m * 16 + khalf * 8 + v;
            float h = acc[m][t8][v] + bias;
            h = 0.5f * h * (1.0f + erff(h * 0.70710678118654752f));
            Xs[row * D + n] = f2bf_bits(h);
        }
        acc[m][t8] = vzero;
    }
    __syncthreads();    // H tile visible to all waves

    // ---- phase 2: OUT = w * (H @ W2[e] + b2[e]) ----
    for (int k0 = 0; k0 < D; k0 += 32) {
        bf16x16 a0 = load_a_frag(Xs, nlo,      khalf, k0);
        bf16x16 a1 = load_a_frag(Xs, nlo + 16, khalf, k0);
        const int kk = k0 + khalf * 8;
#pragma unroll
        for (int t8 = 0; t8 < 8; ++t8) {
            const int n = colbase + t8 * 16 + nlo;
            bf16x16 b = load_b_frag(Wt2e, n, kk);
            acc[0][t8] = WMMA_BF16(a0, b, acc[0][t8]);
            acc[1][t8] = WMMA_BF16(a1, b, acc[1][t8]);
        }
    }

#pragma unroll
    for (int m = 0; m < 2; ++m)
#pragma unroll
    for (int t8 = 0; t8 < 8; ++t8) {
        const int   n    = colbase + t8 * 16 + nlo;
        const float bias = b2[e * D + n];
#pragma unroll
        for (int v = 0; v < 8; ++v) {
            const int row = m * 16 + khalf * 8 + v;
            const int t   = sorted[base + row];
            if (t >= 0) out[(size_t)t * D + n] = gate_w[t] * (acc[m][t8][v] + bias);
        }
    }
}

// ---------------------------------------------------------------- host launcher
extern "C" void kernel_launch(void* const* d_in, const int* in_sizes, int n_in,
                              void* d_out, int out_size, void* d_ws, size_t ws_size,
                              hipStream_t stream)
{
    const float* x  = (const float*)d_in[0];
    const float* Wg = (const float*)d_in[1];
    const float* bg = (const float*)d_in[2];
    const float* W1 = (const float*)d_in[3];
    const float* b1 = (const float*)d_in[4];
    const float* W2 = (const float*)d_in[5];
    const float* b2 = (const float*)d_in[6];
    float* out = (float*)d_out;

    const int NT   = in_sizes[0] / D;        // 8192
    const int MAXT = NT / TM + E;            // padded tile upper bound

    char* ws = (char*)d_ws;
    int*   assign      = (int*)ws;           ws += (size_t)NT * 4;
    float* gatew       = (float*)ws;         ws += (size_t)NT * 4;
    int*   counts      = (int*)ws;           ws += E * 4;
    int*   fill        = (int*)ws;           ws += E * 4;
    int*   seg_base    = (int*)ws;           ws += E * 4;
    int*   n_tiles     = (int*)ws;           ws += 4;
    int*   tile_expert = (int*)ws;           ws += (size_t)MAXT * 4;
    int*   tile_base   = (int*)ws;           ws += (size_t)MAXT * 4;
    int*   sorted      = (int*)ws;           ws += (size_t)(NT + E * TM) * 4;
    unsigned short* Wt1 = (unsigned short*)ws; ws += (size_t)E * D * D * 2;
    unsigned short* Wt2 = (unsigned short*)ws; ws += (size_t)E * D * D * 2;
    (void)n_in; (void)out_size; (void)ws_size;

    hipLaunchKernelGGL(moe_init_kernel, dim3(1), dim3(64), 0, stream, counts, fill);
    hipLaunchKernelGGL(moe_wcvt_kernel, dim3(D / 32, D / 32, E), dim3(32, 8), 0, stream, W1, Wt1);
    hipLaunchKernelGGL(moe_wcvt_kernel, dim3(D / 32, D / 32, E), dim3(32, 8), 0, stream, W2, Wt2);
    hipLaunchKernelGGL(moe_gate_kernel, dim3((unsigned)((NT * 32 + 255) / 256)), dim3(256), 0, stream,
                       x, Wg, bg, assign, gatew, counts, NT);
    hipLaunchKernelGGL(moe_scan_kernel, dim3(1), dim3(1), 0, stream,
                       counts, seg_base, tile_expert, tile_base, n_tiles, sorted);
    hipLaunchKernelGGL(moe_scatter_kernel, dim3((unsigned)((NT + 255) / 256)), dim3(256), 0, stream,
                       assign, seg_base, fill, sorted, NT);
    hipLaunchKernelGGL(moe_mlp_kernel, dim3((unsigned)MAXT), dim3(256), 0, stream,
                       x, Wt1, b1, Wt2, b2, sorted, tile_expert, tile_base, n_tiles, gatew, out);
}
